// GATLSTM_31593779429423
// MI455X (gfx1250) — compile-verified, matrix operand order checked by reading, not run on hardware
//
#include <hip/hip_runtime.h>
#include <stdint.h>

// ---------------- types ----------------
typedef __attribute__((ext_vector_type(16))) __bf16 v16bf;
typedef __attribute__((ext_vector_type(8)))  __bf16 v8bf;
typedef __attribute__((ext_vector_type(8)))  float  v8f;

#define BATCH 64
#define SEQ   512
#define NNODE 30
#define GHID  16          // H*HD = 2*8
#define GDIM  480         // N*GH
#define PDIM  64
#define LH    64
#define NGRAPH (BATCH*SEQ)             // 32768
#define RECON_ELEMS (BATCH*SEQ*NNODE)  // 983040

// workspace layout (bytes, all 16B aligned) -- all activations bf16 now
static constexpr size_t OFF_GAT   = 0;                                  // bf16 [32768][480]
static constexpr size_t SZ_GAT    = (size_t)NGRAPH * GDIM * 2;          // 31457280
static constexpr size_t SZ_SEQB   = (size_t)SEQ * BATCH * PDIM * 2;     // 4194304 (bf16)
static constexpr size_t OFF_PROJB = OFF_GAT + SZ_GAT;                   // bf16 [S][B][64]
static constexpr size_t OFF_H1B   = OFF_PROJB + SZ_SEQB;
static constexpr size_t OFF_H2B   = OFF_H1B + SZ_SEQB;
static constexpr size_t OFF_BPP   = OFF_H2B + SZ_SEQB;                  // bf16 15*4*512
static constexpr size_t OFF_BL0   = OFF_BPP + 30720ull * 2;             // bf16 4*16*512
static constexpr size_t OFF_BL1   = OFF_BL0 + 32768ull * 2;
static constexpr size_t OFF_BO    = OFF_BL1 + 32768ull * 2;             // bf16 2*2*512

// LSTM dynamic-LDS layout (bytes from base of dynamic LDS, which is offset 0)
#define LDS_BW_OFF    0u          // 64 KB  weights (pre-swizzled fragments)
#define LDS_A_OFF     65536u      // 16 KB  A = [x_t | h]  (64 rows x 128 bf16)
#define LDS_GATES_OFF 81920u      // 64 KB  gates f32 (64 x 256)
#define LDS_TOTAL     (65536u + 16384u + 65536u)

// ---------------- CDNA5 async copy (global -> LDS, ASYNCcnt) ----------------
__device__ inline void async_ld_b128_to_lds(uint32_t lds_byte_off, uint64_t gaddr) {
  // GLOBAL_LOAD_ASYNC_TO_LDS_B128: copies 16 B from global memory into LDS at the given byte offset
  asm volatile("global_load_async_to_lds_b128 %0, %1, off"
               :: "v"(lds_byte_off), "v"(gaddr)
               : "memory");
}
__device__ inline void async_wait_all() {
#if __has_builtin(__builtin_amdgcn_s_wait_asynccnt)
  __builtin_amdgcn_s_wait_asynccnt(0);
#else
  asm volatile("s_wait_asynccnt 0x0" ::: "memory");
#endif
}

// ---------------- fragment helpers ----------------
// A-fragment (16x32 bf16): lane<16 holds row M=lane, K {0..7,16..23};
// lane>=16 holds row M=lane-16, K {8..15,24..31}.  Two 16B chunks per lane.
__device__ inline v16bf frag_from_two(const __bf16* p0, const __bf16* p1) {
  v8bf a = *reinterpret_cast<const v8bf*>(p0);
  v8bf b = *reinterpret_cast<const v8bf*>(p1);
  v16bf r;
#pragma unroll
  for (int e = 0; e < 8; ++e) { r[e] = a[e]; r[e + 8] = b[e]; }
  return r;
}

__device__ inline v16bf load_a_frag(const __bf16* row_k0, int half) {
  return frag_from_two(row_k0 + half * 8, row_k0 + 16 + half * 8);
}

// pre-swizzled B fragment: 32 lanes x 16 bf16 contiguous per fragment
__device__ inline v16bf load_b_frag(const __bf16* frag_base, int lane) {
  const __bf16* p = frag_base + lane * 16;
  return frag_from_two(p, p + 8);
}

__device__ inline v8f wmma_bf16(v16bf a, v16bf b, v8f c) {
  return __builtin_amdgcn_wmma_f32_16x16x32_bf16(false, a, false, b, (short)0, c,
                                                 false, false);
}

// ---------------- 1) weight pack / convert ----------------
__global__ void gatlstm_pack_kernel(const float* __restrict__ Wp,
                                    const float* __restrict__ Wih0,
                                    const float* __restrict__ Whh0,
                                    const float* __restrict__ Wih1,
                                    const float* __restrict__ Whh1,
                                    const float* __restrict__ Wout,
                                    unsigned short* bpp_, unsigned short* bl0_,
                                    unsigned short* bl1_, unsigned short* bo_) {
  __bf16* bpp = reinterpret_cast<__bf16*>(bpp_);
  __bf16* bl0 = reinterpret_cast<__bf16*>(bl0_);
  __bf16* bl1 = reinterpret_cast<__bf16*>(bl1_);
  __bf16* bo  = reinterpret_cast<__bf16*>(bo_);
  int idx = blockIdx.x * blockDim.x + threadIdx.x;
  const int NP = 15 * 4 * 512;   // proj:  KT=15 NT=4
  const int NL = 4 * 16 * 512;   // lstm:  KT=4  NT=16
  const int NO = 2 * 2 * 512;    // out:   KT=2  NT=2
  if (idx < NP) {
    int e = idx & 15, lane = (idx >> 4) & 31, frag = idx >> 9;
    int kt = frag >> 2, nt = frag & 3, half = lane >> 4;
    int kl = (e < 8) ? (half * 8 + e) : (16 + half * 8 + (e - 8));
    int k = kt * 32 + kl, n = nt * 16 + (lane & 15);
    bpp[idx] = (__bf16)Wp[k * PDIM + n];
    return;
  }
  idx -= NP;
  if (idx < 2 * NL) {
    const float* Wih = (idx < NL) ? Wih0 : Wih1;
    const float* Whh = (idx < NL) ? Whh0 : Whh1;
    __bf16* dst = (idx < NL) ? bl0 : bl1;
    int j = (idx < NL) ? idx : idx - NL;
    int e = j & 15, lane = (j >> 4) & 31, frag = j >> 9;
    int kt = frag >> 4, nt = frag & 15, half = lane >> 4;
    int kl = (e < 8) ? (half * 8 + e) : (16 + half * 8 + (e - 8));
    int k = kt * 32 + kl, n = nt * 16 + (lane & 15);
    // gates = x@Wih^T + h@Whh^T  -> Bcat[k][n] = k<64 ? Wih[n][k] : Whh[n][k-64]
    float v = (k < LH) ? Wih[n * LH + k] : Whh[n * LH + (k - LH)];
    dst[j] = (__bf16)v;
    return;
  }
  idx -= 2 * NL;
  if (idx < NO) {
    int e = idx & 15, lane = (idx >> 4) & 31, frag = idx >> 9;
    int kt = frag >> 1, nt = frag & 1, half = lane >> 4;
    int kl = (e < 8) ? (half * 8 + e) : (16 + half * 8 + (e - 8));
    int k = kt * 32 + kl, n = nt * 16 + (lane & 15);
    bo[idx] = (__bf16)((n < NNODE) ? Wout[k * NNODE + n] : 0.0f);
  }
}

// ---------------- 2) GAT (one wave32 per graph, lane = node) ----------------
__global__ void gatlstm_gat_kernel(const float* __restrict__ x,
                                   const float* __restrict__ adj,
                                   const float* __restrict__ W_gat,
                                   const float* __restrict__ a_src,
                                   const float* __restrict__ a_dst,
                                   unsigned short* __restrict__ gat_out_,
                                   float* __restrict__ attn_last) {
  __bf16* gat_out = reinterpret_cast<__bf16*>(gat_out_);
  int g = (blockIdx.x * blockDim.x + threadIdx.x) >> 5;   // graph id = b*512+s
  int lane = threadIdx.x & 31;
  if (g >= NGRAPH) return;
  int b = g >> 9, s = g & 511;

  float cs[2], cd[2], wg[2][8];
#pragma unroll
  for (int h = 0; h < 2; ++h) {
    float s_ = 0.f, d_ = 0.f;
#pragma unroll
    for (int d = 0; d < 8; ++d) {
      float w = W_gat[h * 8 + d];
      wg[h][d] = w;
      s_ += w * a_src[h * 8 + d];
      d_ += w * a_dst[h * 8 + d];
    }
    cs[h] = s_; cd[h] = d_;
  }

  int i = lane;
  float xi = (i < NNODE) ? x[(size_t)g * NNODE + i] : 0.f;
  float es0 = cs[0] * xi, es1 = cs[1] * xi;

  // pass 1: row max (adj has self loops -> row never empty)
  float m0 = -1e30f, m1 = -1e30f;
  for (int j = 0; j < NNODE; ++j) {
    float xj = __shfl(xi, j, 32);
    float a = (i < NNODE) ? adj[i * NNODE + j] : 0.f;
    if (a != 0.f) {
      float e0 = es0 + cd[0] * xj; e0 = (e0 >= 0.f) ? e0 : 0.2f * e0;
      float e1 = es1 + cd[1] * xj; e1 = (e1 >= 0.f) ? e1 : 0.2f * e1;
      m0 = fmaxf(m0, e0); m1 = fmaxf(m1, e1);
    }
  }
  // pass 2: softmax denom + weighted sum of x_j
  float sum0 = 0.f, sum1 = 0.f, w0 = 0.f, w1 = 0.f;
  for (int j = 0; j < NNODE; ++j) {
    float xj = __shfl(xi, j, 32);
    float a = (i < NNODE) ? adj[i * NNODE + j] : 0.f;
    if (a != 0.f) {
      float e0 = es0 + cd[0] * xj; e0 = (e0 >= 0.f) ? e0 : 0.2f * e0;
      float e1 = es1 + cd[1] * xj; e1 = (e1 >= 0.f) ? e1 : 0.2f * e1;
      float p0 = __expf(e0 - m0), p1 = __expf(e1 - m1);
      sum0 += p0; sum1 += p1; w0 += p0 * xj; w1 += p1 * xj;
    }
  }
  float inv0 = (sum0 > 0.f) ? 1.f / sum0 : 0.f;
  float inv1 = (sum1 > 0.f) ? 1.f / sum1 : 0.f;
  float sx0 = w0 * inv0, sx1 = w1 * inv1;

  if (i < NNODE) {
    v8bf o0, o1;
#pragma unroll
    for (int d = 0; d < 8; ++d) {
      float v0 = sx0 * wg[0][d]; v0 = (v0 > 0.f) ? v0 : (__expf(v0) - 1.f);
      float v1 = sx1 * wg[1][d]; v1 = (v1 > 0.f) ? v1 : (__expf(v1) - 1.f);
      o0[d] = (__bf16)v0; o1[d] = (__bf16)v1;
    }
    __bf16* dst = gat_out + (size_t)g * GDIM + (size_t)i * GHID;
    *reinterpret_cast<v8bf*>(dst)     = o0;
    *reinterpret_cast<v8bf*>(dst + 8) = o1;

    if (s == SEQ - 1) {  // head-averaged attention row for the last timestep
      float* arow = attn_last + (size_t)b * (NNODE * NNODE) + (size_t)i * NNODE;
      for (int j = 0; j < NNODE; ++j) {
        float xj = __shfl(xi, j, 32);
        float a = adj[i * NNODE + j];
        float v = 0.f;
        if (a != 0.f) {
          float e0 = es0 + cd[0] * xj; e0 = (e0 >= 0.f) ? e0 : 0.2f * e0;
          float e1 = es1 + cd[1] * xj; e1 = (e1 >= 0.f) ? e1 : 0.2f * e1;
          v = 0.5f * (__expf(e0 - m0) * inv0 + __expf(e1 - m1) * inv1);
        }
        arow[j] = v;
      }
    }
  }
}

// ---------------- 3) projection GEMM: relu((32768x480)@(480x64)+b) ----------
// block = 4 waves; wave w computes C tile rows [rt*16, +16), cols [w*16, +16)
// output stored bf16 in (s, b, p) layout for async staging into the LSTM.
__global__ void gatlstm_proj_kernel(const unsigned short* __restrict__ A_,
                                    const unsigned short* __restrict__ Bp_,
                                    const float* __restrict__ bias,
                                    unsigned short* __restrict__ out_) {
  const __bf16* A  = reinterpret_cast<const __bf16*>(A_);
  const __bf16* Bp = reinterpret_cast<const __bf16*>(Bp_);
  __bf16* out = reinterpret_cast<__bf16*>(out_);
  int rt = blockIdx.x;                 // 0..2047
  int wv = threadIdx.x >> 5;           // 0..3 (col tile)
  int lane = threadIdx.x & 31;
  int half = lane >> 4, r = lane & 15;
  const __bf16* arow = A + (size_t)(rt * 16 + r) * GDIM;
  v8f acc = {};
#pragma unroll 3
  for (int kt = 0; kt < 15; ++kt) {
    if (kt + 1 < 15) __builtin_prefetch(arow + (kt + 1) * 32, 0, 3);  // near-scope
    v16bf af = load_a_frag(arow + kt * 32, half);
    v16bf bf = load_b_frag(Bp + (size_t)(kt * 4 + wv) * 512, lane);
    acc = wmma_bf16(af, bf, acc);
  }
  int n = wv * 16 + r;
  float bb = bias[n];
#pragma unroll
  for (int k = 0; k < 8; ++k) {
    int g = rt * 16 + half * 8 + k;          // g = b*512 + s
    int b_ = g >> 9, s_ = g & 511;
    float v = acc[k] + bb;
    out[((size_t)(s_ * BATCH + b_)) * PDIM + n] = (__bf16)((v > 0.f) ? v : 0.f);
  }
}

// ---------------- 4) persistent LSTM layer (1 workgroup, 8 waves) ----------
// per step: gates(64x256) = [x_t|h](64x128) @ Bcat(128x256); weights in LDS.
// x_t staged global->LDS with CDNA5 async copies (ASYNCcnt).
__global__ void gatlstm_lstm_kernel(const unsigned short* __restrict__ xin_,
                                    const unsigned short* __restrict__ Bp_,
                                    const float* __restrict__ b_ih,
                                    const float* __restrict__ b_hh,
                                    unsigned short* __restrict__ hout_) {
  extern __shared__ char smem[];
  __bf16* Bw    = reinterpret_cast<__bf16*>(smem + LDS_BW_OFF);     // 64 KB weights
  __bf16* Aw    = reinterpret_cast<__bf16*>(smem + LDS_A_OFF);      // 16 KB [64][128]
  float*  gates = reinterpret_cast<float*>(smem + LDS_GATES_OFF);   // 64 KB [64][256]
  __bf16* hout  = reinterpret_cast<__bf16*>(hout_);
  int t = threadIdx.x;                 // 256 threads = 8 waves
  int lane = t & 31, wv = t >> 5;
  int half = lane >> 4, r = lane & 15;

  { // stage weights into LDS once (4096 x 16B)
    const uint4* src = reinterpret_cast<const uint4*>(Bp_);
    uint4* dst = reinterpret_cast<uint4*>(Bw);
    for (int i = t; i < 4096; i += 256) dst[i] = src[i];
  }
  for (int i = t; i < BATCH * LH; i += 256) { // h0 = 0
    int b_ = i >> 6, u = i & 63;
    Aw[b_ * 128 + LH + u] = (__bf16)0.f;
  }
  float creg[16];
#pragma unroll
  for (int j = 0; j < 16; ++j) creg[j] = 0.f;   // c0 = 0
  __syncthreads();

  int mt = wv >> 1;                 // 4 row tiles, 2 waves each
  int ntbase = (wv & 1) * 8;        // 8 col tiles per wave
  int m0 = mt * 16;

  for (int s = 0; s < SEQ; ++s) {
    // phase 1: async-copy x_t (64 rows x 128 B) into the x-half of A.
    // 512 16-byte chunks; chunk c -> row c>>3, 16B-chunk (c&7) of that row.
    {
      uint64_t gbase = (uint64_t)(uintptr_t)xin_ + (uint64_t)s * (BATCH * PDIM * 2);
#pragma unroll
      for (int rep = 0; rep < 2; ++rep) {
        int c = t + rep * 256;
        uint32_t lds_off = LDS_A_OFF + (uint32_t)(c >> 3) * 256u + (uint32_t)(c & 7) * 16u;
        async_ld_b128_to_lds(lds_off, gbase + (uint64_t)c * 16u);
      }
      async_wait_all();
    }
    __syncthreads();

    // phase 2: WMMA GEMM -> gates
    v16bf af[4];
#pragma unroll
    for (int kt = 0; kt < 4; ++kt)
      af[kt] = load_a_frag(Aw + (m0 + r) * 128 + kt * 32, half);
#pragma unroll
    for (int q = 0; q < 8; ++q) {
      int nt = ntbase + q;
      v8f acc = {};
#pragma unroll
      for (int kt = 0; kt < 4; ++kt) {
        v16bf bf = load_b_frag(Bw + (size_t)(kt * 16 + nt) * 512, lane);
        acc = wmma_bf16(af[kt], bf, acc);
      }
      int n = nt * 16 + r;
#pragma unroll
      for (int k = 0; k < 8; ++k)
        gates[(m0 + half * 8 + k) * 256 + n] = acc[k];
    }
    __syncthreads();

    // phase 3: gate nonlinearities, state update, h -> A (bf16) and global (bf16)
    __bf16* hrow = hout + (size_t)s * (BATCH * LH);
#pragma unroll
    for (int j = 0; j < 16; ++j) {
      int idx = t * 16 + j;
      int b_ = idx >> 6, u = idx & 63;
      const float* gr = gates + b_ * 256;
      float gi = gr[u]       + b_ih[u]       + b_hh[u];
      float gf = gr[64 + u]  + b_ih[64 + u]  + b_hh[64 + u];
      float gg = gr[128 + u] + b_ih[128 + u] + b_hh[128 + u];
      float go = gr[192 + u] + b_ih[192 + u] + b_hh[192 + u];
      float si = 1.f / (1.f + __expf(-gi));
      float sf = 1.f / (1.f + __expf(-gf));
      float so = 1.f / (1.f + __expf(-go));
      float tg = tanhf(gg);
      float c = sf * creg[j] + si * tg;
      creg[j] = c;
      float h = so * tanhf(c);
      __bf16 hb = (__bf16)h;
      Aw[b_ * 128 + LH + u] = hb;
      hrow[idx] = hb;
    }
    __syncthreads();
  }
}

// ---------------- 5) output GEMM: (32768x64)@(64x32pad) + b ------------------
__global__ void gatlstm_out_kernel(const unsigned short* __restrict__ h2_,
                                   const unsigned short* __restrict__ Bp_,
                                   const float* __restrict__ b_out,
                                   float* __restrict__ recon) {
  const __bf16* h2 = reinterpret_cast<const __bf16*>(h2_);
  const __bf16* Bp = reinterpret_cast<const __bf16*>(Bp_);
  int rt = blockIdx.x;                 // rows in (s,b) order
  int wv = threadIdx.x >> 5;           // 0..1 col tile
  int lane = threadIdx.x & 31;
  int half = lane >> 4, r = lane & 15;
  const __bf16* arow = h2 + (size_t)(rt * 16 + r) * LH;
  v8f acc = {};
#pragma unroll
  for (int kt = 0; kt < 2; ++kt) {
    v16bf af = load_a_frag(arow + kt * 32, half);
    v16bf bf = load_b_frag(Bp + (size_t)(kt * 2 + wv) * 512, lane);
    acc = wmma_bf16(af, bf, acc);
  }
  int n = wv * 16 + r;
  if (n < NNODE) {
    float bb = b_out[n];
#pragma unroll
    for (int k = 0; k < 8; ++k) {
      int g = rt * 16 + half * 8 + k;        // g = s*64 + b
      int s_ = g >> 6, b_ = g & 63;
      recon[((size_t)b_ * SEQ + s_) * NNODE + n] = acc[k] + bb;
    }
  }
}

// ---------------- launch ----------------
extern "C" void kernel_launch(void* const* d_in, const int* in_sizes, int n_in,
                              void* d_out, int out_size, void* d_ws, size_t ws_size,
                              hipStream_t stream) {
  const float* x      = (const float*)d_in[0];
  const float* adj    = (const float*)d_in[1];
  const float* W_gat  = (const float*)d_in[2];
  const float* a_src  = (const float*)d_in[3];
  const float* a_dst  = (const float*)d_in[4];
  const float* W_proj = (const float*)d_in[5];
  const float* b_proj = (const float*)d_in[6];
  const float* W_ih0  = (const float*)d_in[7];
  const float* W_hh0  = (const float*)d_in[8];
  const float* b_ih0  = (const float*)d_in[9];
  const float* b_hh0  = (const float*)d_in[10];
  const float* W_ih1  = (const float*)d_in[11];
  const float* W_hh1  = (const float*)d_in[12];
  const float* b_ih1  = (const float*)d_in[13];
  const float* b_hh1  = (const float*)d_in[14];
  const float* W_out  = (const float*)d_in[15];
  const float* b_out  = (const float*)d_in[16];

  float* out = (float*)d_out;
  char* ws = (char*)d_ws;
  unsigned short* gatb  = (unsigned short*)(ws + OFF_GAT);
  unsigned short* projb = (unsigned short*)(ws + OFF_PROJB);
  unsigned short* h1b   = (unsigned short*)(ws + OFF_H1B);
  unsigned short* h2b   = (unsigned short*)(ws + OFF_H2B);
  unsigned short* bpp   = (unsigned short*)(ws + OFF_BPP);
  unsigned short* bl0   = (unsigned short*)(ws + OFF_BL0);
  unsigned short* bl1   = (unsigned short*)(ws + OFF_BL1);
  unsigned short* bo    = (unsigned short*)(ws + OFF_BO);

  gatlstm_pack_kernel<<<384, 256, 0, stream>>>(W_proj, W_ih0, W_hh0, W_ih1, W_hh1,
                                               W_out, bpp, bl0, bl1, bo);
  gatlstm_gat_kernel<<<NGRAPH / 8, 256, 0, stream>>>(x, adj, W_gat, a_src, a_dst,
                                                     gatb, out + RECON_ELEMS);
  gatlstm_proj_kernel<<<NGRAPH / 16, 128, 0, stream>>>(gatb, bpp, b_proj, projb);
  gatlstm_lstm_kernel<<<1, 256, LDS_TOTAL, stream>>>(projb, bl0, b_ih0, b_hh0, h1b);
  gatlstm_lstm_kernel<<<1, 256, LDS_TOTAL, stream>>>(h1b, bl1, b_ih1, b_hh1, h2b);
  gatlstm_out_kernel<<<NGRAPH / 16, 64, 0, stream>>>(h2b, bo, b_out, out);
}
